// LSTMlayer_35871566856645
// MI455X (gfx1250) — compile-verified, matrix operand order checked by reading, not run on hardware
//
#include <hip/hip_runtime.h>
#include <stdint.h>

// ---------------------------------------------------------------------------
// LSTM on MI455X (gfx1250, wave32, WMMA).
// Per step: gates = [x_t | h] @ [Wx ; Wh] + b == [128 x 1536] @ [1536 x 4096]
// bf16 WMMA (v_wmma_f32_16x16x32_bf16) with f32 accumulation.
// Weights converted once to bf16, transposed [n][k]; 12.6 MB -> L2-resident
// (192 MB L2) across all 512 steps. h-state double-buffered in bf16.
// LDS B-tile ping/pong: global fill of chunk k+1 overlaps the 8 WMMAs of
// chunk k; one barrier per 64-k chunk.
// Workspace: ~80.3 MB (Wt 12.6MB + x_bf16 64MB + 2 x h_bf16 256KB).
// ---------------------------------------------------------------------------

#define BATCH  128
#define SEQ    512
#define DIN    512
#define UNITS  1024
#define KTOT   (DIN + UNITS)      // 1536
#define KCHUNK 64                 // k per pipelined iteration
#define NITER  (KTOT / KCHUNK)    // 24
#define LROW   72                 // padded LDS row stride in u16 (144B, 16B-mult)

typedef __attribute__((ext_vector_type(16))) __bf16 v16bf;
typedef __attribute__((ext_vector_type(8)))  __bf16 v8bf;
typedef __attribute__((ext_vector_type(8)))  float  v8f;

static __device__ __forceinline__ uint16_t f32_to_bf16(float f) {
    uint32_t u = __float_as_uint(f);
    uint32_t r = u + 0x7FFFu + ((u >> 16) & 1u);   // round-to-nearest-even
    return (uint16_t)(r >> 16);
}

static __device__ __forceinline__ float sigm(float x) {
    return 1.0f / (1.0f + __expf(-x));
}

static __device__ __forceinline__ v16bf cat16(v8bf lo, v8bf hi) {
    return __builtin_shufflevector(lo, hi, 0, 1, 2, 3, 4, 5, 6, 7,
                                           8, 9, 10, 11, 12, 13, 14, 15);
}

// --- one-time converts -----------------------------------------------------

// Wt[n][k] = bf16( k < DIN ? Wx[k][n] : Wh[k-DIN][n] ),  n in [0,4096)
__global__ __launch_bounds__(256) void cvt_weights(const float* __restrict__ Wx,
                                                   const float* __restrict__ Wh,
                                                   uint16_t* __restrict__ Wt) {
    const int k = blockIdx.x * 256 + threadIdx.x;   // gridDim.x = 6 -> k < 1536
    const int n = blockIdx.y;                       // gridDim.y = 4096
    if (k >= KTOT) return;
    float v = (k < DIN) ? Wx[(size_t)k * 4096 + n]
                        : Wh[(size_t)(k - DIN) * 4096 + n];
    Wt[(size_t)n * KTOT + k] = f32_to_bf16(v);
}

__global__ __launch_bounds__(256) void cvt_x(const float* __restrict__ x,
                                             uint16_t* __restrict__ xq) {
    const size_t i = (size_t)blockIdx.x * 256 + threadIdx.x;
    xq[i] = f32_to_bf16(x[i]);
}

__global__ __launch_bounds__(256) void init_state(float* __restrict__ c_state,
                                                  uint16_t* __restrict__ hq0,
                                                  uint16_t* __restrict__ hq1) {
    const int i = blockIdx.x * 256 + threadIdx.x;   // grid covers 131072
    c_state[i] = 0.0f;
    hq0[i] = 0;
    hq1[i] = 0;
}

// --- one LSTM timestep -----------------------------------------------------
// grid: 64 WGs (one per 16-unit slice), 256 threads = 8 waves (wave w = rows 16w..)
__global__ __launch_bounds__(256)
void lstm_step(const uint16_t* __restrict__ xq,      // [B][S][DIN] bf16
               const uint16_t* __restrict__ Wt,      // [4096][1536] bf16
               const float*    __restrict__ bias,    // [4096]
               const uint16_t* __restrict__ h_in,    // [B][U] bf16 (state t-1)
               uint16_t*       __restrict__ h_nq,    // [B][U] bf16 (state t)
               float*          __restrict__ h_out,   // [B][U] f32 (d_out)
               float*          __restrict__ c_state, // [B][U] f32 (d_out half 2)
               int t)
{
    // ping/pong B tiles: 64 rows (gate*16 + n) x 64 k, padded stride
    __shared__ __align__(16) uint16_t Bs[2][64 * LROW];   // 2 x 9 KB

    const int tid  = threadIdx.x;
    const int lane = tid & 31;
    const int wave = tid >> 5;
    const int u0   = blockIdx.x * 16;       // unit slice
    const int m0   = wave * 16;             // batch-row tile
    const int nl   = lane & 15;
    const int half = lane >> 4;             // 0: lanes 0-15, 1: lanes 16-31
    const int kb   = half * 8;              // ISA 16-bit A/B per-half K offset

    // cooperative fill: 64 rows x 8 granules(16B) = 512 granules, 2 per thread
    const int r0 = tid >> 3,          s0 = tid & 7;            // granule tid
    const int r1 = (tid + 256) >> 3,  s1 = tid & 7;            // granule tid+256
    const uint16_t* bsrc0 = Wt + (size_t)(u0 + (r0 & 15) + (r0 >> 4) * UNITS) * KTOT;
    const uint16_t* bsrc1 = Wt + (size_t)(u0 + (r1 & 15) + (r1 >> 4) * UNITS) * KTOT;
    const int loff0 = r0 * LROW + s0 * 8;
    const int loff1 = r1 * LROW + s1 * 8;

    // per-lane A rows
    const int m_a = m0 + nl;
    const uint16_t* xrow = xq + ((size_t)m_a * SEQ + t) * DIN;
    const uint16_t* hrow = h_in + (size_t)m_a * UNITS;

    v8f acc0 = {}, acc1 = {}, acc2 = {}, acc3 = {};

    // prologue: fill buffer 0 with chunk 0
    v8bf nb0 = *(const v8bf*)&bsrc0[s0 * 8];
    v8bf nb1 = *(const v8bf*)&bsrc1[s1 * 8];
    *(v8bf*)&Bs[0][loff0] = nb0;
    *(v8bf*)&Bs[0][loff1] = nb1;
    __syncthreads();

    for (int kc = 0; kc < NITER; ++kc) {
        const int k0  = kc * KCHUNK;
        const int cur = kc & 1;

        // issue next chunk's global loads early (latency hides under WMMAs)
        if (kc + 1 < NITER) {
            nb0 = *(const v8bf*)&bsrc0[k0 + KCHUNK + s0 * 8];
            nb1 = *(const v8bf*)&bsrc1[k0 + KCHUNK + s1 * 8];
        }

        // A region is uniform per chunk (512 is a multiple of KCHUNK)
        const uint16_t* arow = (k0 < DIN) ? (xrow + k0) : (hrow + (k0 - DIN));

        #pragma unroll
        for (int c = 0; c < 2; ++c) {          // two 32-k sub-chunks
            const uint16_t* ar = arow + c * 32;
            v16bf av = cat16(*(const v8bf*)&ar[kb], *(const v8bf*)&ar[kb + 16]);

            // load all 4 gate B-operands first, then burst 4 WMMAs
            const uint16_t* base = &Bs[cur][c * 32 + kb];
            v16bf bv0 = cat16(*(const v8bf*)&base[(0 * 16 + nl) * LROW],
                              *(const v8bf*)&base[(0 * 16 + nl) * LROW + 16]);
            v16bf bv1 = cat16(*(const v8bf*)&base[(1 * 16 + nl) * LROW],
                              *(const v8bf*)&base[(1 * 16 + nl) * LROW + 16]);
            v16bf bv2 = cat16(*(const v8bf*)&base[(2 * 16 + nl) * LROW],
                              *(const v8bf*)&base[(2 * 16 + nl) * LROW + 16]);
            v16bf bv3 = cat16(*(const v8bf*)&base[(3 * 16 + nl) * LROW],
                              *(const v8bf*)&base[(3 * 16 + nl) * LROW + 16]);

            acc0 = __builtin_amdgcn_wmma_f32_16x16x32_bf16(
                false, av, false, bv0, (short)0, acc0, false, false);
            acc1 = __builtin_amdgcn_wmma_f32_16x16x32_bf16(
                false, av, false, bv1, (short)0, acc1, false, false);
            acc2 = __builtin_amdgcn_wmma_f32_16x16x32_bf16(
                false, av, false, bv2, (short)0, acc2, false, false);
            acc3 = __builtin_amdgcn_wmma_f32_16x16x32_bf16(
                false, av, false, bv3, (short)0, acc3, false, false);
        }

        // store next chunk into the other buffer, then one barrier
        if (kc + 1 < NITER) {
            *(v8bf*)&Bs[cur ^ 1][loff0] = nb0;
            *(v8bf*)&Bs[cur ^ 1][loff1] = nb1;
        }
        __syncthreads();
    }

    // Epilogue: C/D layout -> all 4 gates of (m,u) live in the same lane/slot.
    const int u = u0 + nl;
    const float bi = bias[u];
    const float bf = bias[UNITS + u];
    const float bg = bias[2 * UNITS + u];
    const float bo = bias[3 * UNITS + u];

    #pragma unroll
    for (int r = 0; r < 8; ++r) {
        const int m = m0 + r + half * 8;       // C layout: lanes>=16 -> M=r+8
        const size_t idx = (size_t)m * UNITS + u;
        const float gi = sigm(acc0[r] + bi);
        const float gf = sigm(acc1[r] + bf);
        const float gg = tanhf(acc2[r] + bg);
        const float go = sigm(acc3[r] + bo);
        const float c  = gf * c_state[idx] + gi * gg;
        const float h  = go * tanhf(c);
        c_state[idx] = c;
        h_out[idx]   = h;
        h_nq[idx]    = f32_to_bf16(h);
    }
}

// ---------------------------------------------------------------------------

extern "C" void kernel_launch(void* const* d_in, const int* in_sizes, int n_in,
                              void* d_out, int out_size, void* d_ws, size_t ws_size,
                              hipStream_t stream) {
    const float* x  = (const float*)d_in[0];   // [128][512][512]
    const float* Wx = (const float*)d_in[1];   // [512][4096]
    const float* Wh = (const float*)d_in[2];   // [1024][4096]
    const float* b  = (const float*)d_in[3];   // [4096]

    float* out     = (float*)d_out;
    float* h_out   = out;                      // [128][1024]
    float* c_state = out + BATCH * UNITS;      // [128][1024], doubles as c state

    uint8_t* ws = (uint8_t*)d_ws;
    uint16_t* Wt  = (uint16_t*)(ws);                        // 12,582,912 B
    uint16_t* xq  = (uint16_t*)(ws + 12582912);             // 67,108,864 B
    uint16_t* hq0 = (uint16_t*)(ws + 12582912 + 67108864);  //    262,144 B
    uint16_t* hq1 = (uint16_t*)(ws + 12582912 + 67108864 + 262144);

    cvt_weights<<<dim3(6, 4096), 256, 0, stream>>>(Wx, Wh, Wt);
    cvt_x<<<(BATCH * SEQ * DIN) / 256, 256, 0, stream>>>(x, xq);
    init_state<<<(BATCH * UNITS) / 256, 256, 0, stream>>>(c_state, hq0, hq1);

    for (int t = 0; t < SEQ; ++t) {
        uint16_t* hin  = (t & 1) ? hq1 : hq0;
        uint16_t* hnew = (t & 1) ? hq0 : hq1;
        lstm_step<<<UNITS / 16, 256, 0, stream>>>(xq, Wt, b, hin, hnew,
                                                  h_out, c_state, t);
    }
}
